// AutoEncoder_7499012899672
// MI455X (gfx1250) — matrix operand order repaired and checked by reading, compiled, not run on hardware
//
#include <hip/hip_runtime.h>
#include <cstdint>
#include <cstddef>

#define B_DIM 4096
#define D_DIM 1024
#define F_DIM 24576
#define TOPK 32
#define DEAD_TOPK 256
#define DEAD_CUTOFF 100000
#define FUZZ_FACTOR 1.0f

// ---------- types ----------
typedef __attribute__((ext_vector_type(16))) __bf16          v16bf;
typedef __attribute__((ext_vector_type(8)))  float           v8f;
typedef __attribute__((ext_vector_type(4)))  unsigned short  us4;
typedef __attribute__((ext_vector_type(8)))  unsigned short  us8;
typedef __attribute__((ext_vector_type(16))) unsigned short  us16;
typedef __attribute__((ext_vector_type(4)))  unsigned int    v4u;
typedef __attribute__((ext_vector_type(8)))  int             v8i;
typedef __attribute__((ext_vector_type(4)))  int             v4i;

__device__ __forceinline__ unsigned f2u(float x) { union { float f; unsigned u; } c; c.f = x; return c.u; }
__device__ __forceinline__ float u2f(unsigned u) { union { float f; unsigned u; } c; c.u = u; return c.f; }

// round-to-nearest-even fp32 -> bf16 (bits)
__device__ __forceinline__ unsigned short bf_rne(float x) {
    unsigned u = f2u(x);
    return (unsigned short)((u + 0x7FFFu + ((u >> 16) & 1u)) >> 16);
}
// split fp32 into hi/lo bf16 halves: x ~= hi + lo with ~16 mantissa bits
__device__ __forceinline__ void split_bf(float x, unsigned short& h, unsigned short& l) {
    unsigned short hb = bf_rne(x);
    float fh = u2f((unsigned)hb << 16);
    h = hb;
    l = bf_rne(x - fh);
}

// =====================================================================
// Kernel 0: fp32 -> (hi, lo) bf16 split, optional bias subtract
// =====================================================================
__global__ __launch_bounds__(256)
void split_kernel(const float* __restrict__ src, const float* __restrict__ bias,
                  unsigned short* __restrict__ hi, unsigned short* __restrict__ lo,
                  int rows)
{
    const size_t n4 = (size_t)rows * (D_DIM / 4);
    const size_t stride = (size_t)gridDim.x * blockDim.x;
    for (size_t q = (size_t)blockIdx.x * blockDim.x + threadIdx.x; q < n4; q += stride) {
        const float4 v = ((const float4*)src)[q];
        float x[4] = { v.x, v.y, v.z, v.w };
        if (bias) {
            int c4 = (int)(q & (D_DIM / 4 - 1)) * 4;
            const float4 b = *(const float4*)(bias + c4);
            x[0] -= b.x; x[1] -= b.y; x[2] -= b.z; x[3] -= b.w;
        }
        us4 h, l;
        #pragma unroll
        for (int j = 0; j < 4; ++j) { unsigned short hh, ll; split_bf(x[j], hh, ll); h[j] = hh; l[j] = ll; }
        ((us4*)hi)[q] = h;
        ((us4*)lo)[q] = l;
    }
}

// =====================================================================
// Kernel 1: project = X @ W^T via split-bf16 WMMA, TDM double-buffered
// =====================================================================
#define BM 128
#define BN 128
#define BK 32
#define LDT 40                   // LDS row stride (elements): 32 data + 8 pad (TDM pad: 4 DWORDs)
#define TILE_E (BM * LDT)        // 5120 elements per tile
#define BUF_E  (4 * TILE_E)      // Ahi, Alo, Bhi, Blo
#define GEMM_SMEM_BYTES (2 * BUF_E * 2)   // 81920 B, double buffered

#if __has_builtin(__builtin_amdgcn_tensor_load_to_lds)
#define HAVE_TDM 1
#else
#define HAVE_TDM 0
#endif

#if HAVE_TDM
// Issue one TDM tile load: 128 rows x 32 bf16, row pitch D_DIM elements in
// global, padded to LDT elements in LDS (pad_interval=16 DW, pad_amount=4 DW).
__device__ __forceinline__ void tdm_issue(int which, int buf, int k0, int m0, int f0,
                                          const unsigned short* __restrict__ Xhi,
                                          const unsigned short* __restrict__ Xlo,
                                          const unsigned short* __restrict__ Whi,
                                          const unsigned short* __restrict__ Wlo,
                                          unsigned smem_base)
{
    const unsigned short* gp;
    unsigned rows;
    if (which == 0)      { gp = Xhi + (size_t)m0 * D_DIM + k0; rows = B_DIM; }
    else if (which == 1) { gp = Xlo + (size_t)m0 * D_DIM + k0; rows = B_DIM; }
    else if (which == 2) { gp = Whi + (size_t)f0 * D_DIM + k0; rows = F_DIM; }
    else                 { gp = Wlo + (size_t)f0 * D_DIM + k0; rows = F_DIM; }

    const unsigned lds = smem_base + (unsigned)((buf * 4 + which) * TILE_E) * 2u;
    const unsigned long long ga = (unsigned long long)(size_t)gp;

    // D# group 0: count=1 | lds_addr | global_addr[56:0] | type=2
    v4u g0;
    g0[0] = 1u;
    g0[1] = lds;
    g0[2] = (unsigned)ga;
    g0[3] = (unsigned)((ga >> 32) & 0x01FFFFFFull) | (2u << 30);
    // D# group 1
    v8i g1;
    g1[0] = (int)((1u << 16) | (1u << 20) | (3u << 22) | (3u << 25)); // data_size=2B, pad_en, 16DW interval, 4DW pad
    g1[1] = (int)(((unsigned)D_DIM & 0xFFFFu) << 16);                 // tensor_dim0 lo16 @ [63:48]
    g1[2] = (int)((((unsigned)D_DIM >> 16) & 0xFFFFu) | ((rows & 0xFFFFu) << 16)); // dim0 hi16 | dim1 lo16
    g1[3] = (int)(((rows >> 16) & 0xFFFFu) | (32u << 16));            // dim1 hi16 | tile_dim0=32
    g1[4] = (int)128;                                                 // tile_dim1=128, tile_dim2=0
    g1[5] = (int)D_DIM;                                               // tensor_dim0_stride lo32
    g1[6] = 0;
    g1[7] = 0;
    v4i g2 = { 0, 0, 0, 0 };
    v4i g3 = { 0, 0, 0, 0 };
    v8i g4 = { 0, 0, 0, 0, 0, 0, 0, 0 };   // 6-arg toolchain: extra group, zeros for 2-D tile
    __builtin_amdgcn_tensor_load_to_lds(g0, g1, g2, g3, g4, 0);
}
#endif

// One K-step of the split-bf16 WMMA compute (24 wmma ops per wave).
__device__ __forceinline__ void wmma_step(const unsigned short* __restrict__ As_hi,
                                          const unsigned short* __restrict__ As_lo,
                                          const unsigned short* __restrict__ Bs_hi,
                                          const unsigned short* __restrict__ Bs_lo,
                                          int lane, int wm, int wn, v8f acc[2][4])
{
    // B fragments: operand B = W_tile^T. B-layout (16-bit, 32x16):
    // lane n<16 holds col n, K {0..7, 16..23} ; lane n+16 holds col n, K {8..15, 24..31}
    v16bf bh[4], bl[4];
    #pragma unroll
    for (int ni = 0; ni < 4; ++ni) {
        int brow = wn + ni * 16 + (lane & 15);
        int bc   = (lane < 16) ? 0 : 8;
        union { us8 h[2]; us16 w; } uh, ul;
        uh.h[0] = *(const us8*)(Bs_hi + brow * LDT + bc);
        uh.h[1] = *(const us8*)(Bs_hi + brow * LDT + bc + 16);
        ul.h[0] = *(const us8*)(Bs_lo + brow * LDT + bc);
        ul.h[1] = *(const us8*)(Bs_lo + brow * LDT + bc + 16);
        bh[ni] = __builtin_bit_cast(v16bf, uh.w);
        bl[ni] = __builtin_bit_cast(v16bf, ul.w);
    }
    // A fragments: A-layout (16-bit, 16x32):
    // lane m<16: row m, K {0..7, 16..23}; lane m+16: row m, K {8..15, 24..31}
    #pragma unroll
    for (int mi = 0; mi < 2; ++mi) {
        int arow = wm + mi * 16 + (lane & 15);
        int c0   = (lane < 16) ? 0 : 8;
        union { us8 h[2]; us16 w; } ch, cl;
        ch.h[0] = *(const us8*)(As_hi + arow * LDT + c0);
        ch.h[1] = *(const us8*)(As_hi + arow * LDT + c0 + 16);
        cl.h[0] = *(const us8*)(As_lo + arow * LDT + c0);
        cl.h[1] = *(const us8*)(As_lo + arow * LDT + c0 + 16);
        v16bf ah = __builtin_bit_cast(v16bf, ch.w);
        v16bf al = __builtin_bit_cast(v16bf, cl.w);
        #pragma unroll
        for (int ni = 0; ni < 4; ++ni) {
            // (hi+lo)*(hi+lo) ~= hi*hi + hi*lo + lo*hi (lo*lo ~ 2^-32, dropped)
            acc[mi][ni] = __builtin_amdgcn_wmma_f32_16x16x32_bf16(
                false, ah, false, bh[ni], (short)0, acc[mi][ni], false, false);
            acc[mi][ni] = __builtin_amdgcn_wmma_f32_16x16x32_bf16(
                false, ah, false, bl[ni], (short)0, acc[mi][ni], false, false);
            acc[mi][ni] = __builtin_amdgcn_wmma_f32_16x16x32_bf16(
                false, al, false, bh[ni], (short)0, acc[mi][ni], false, false);
        }
    }
}

__global__ __launch_bounds__(256)
void gemm_project_kernel(const unsigned short* __restrict__ Xhi,
                         const unsigned short* __restrict__ Xlo,
                         const unsigned short* __restrict__ Whi,
                         const unsigned short* __restrict__ Wlo,
                         float* __restrict__ project,
                         double* __restrict__ blocksums)
{
    extern __shared__ unsigned short smem[];   // 2 buffers x 4 tiles x TILE_E

    const int tid  = threadIdx.x;
    const int lane = tid & 31;
    const int wv   = tid >> 5;
    const int wm   = (wv & 3) * 32;
    const int wn   = (wv >> 2) * 64;
    const int m0   = blockIdx.y * BM;
    const int f0   = blockIdx.x * BN;
    const int nk   = D_DIM / BK;

    v8f acc[2][4] = {};

#if HAVE_TDM
    const int wvu = __builtin_amdgcn_readfirstlane(wv);
    const unsigned smem_base = (unsigned)(size_t)(void*)smem;

    if (wvu < 4) tdm_issue(wvu, 0, 0, m0, f0, Xhi, Xlo, Whi, Wlo, smem_base);

    for (int kk = 0; kk < nk; ++kk) {
        const int cur = kk & 1;
        if (wvu < 4) __builtin_amdgcn_s_wait_tensorcnt(0);
        __syncthreads();                                  // buf[cur] ready; prev compute done
        if (kk + 1 < nk && wvu < 4)
            tdm_issue(wvu, 1 - cur, (kk + 1) * BK, m0, f0, Xhi, Xlo, Whi, Wlo, smem_base);
        wmma_step(smem + (size_t)cur * BUF_E,
                  smem + (size_t)cur * BUF_E + TILE_E,
                  smem + (size_t)cur * BUF_E + 2 * TILE_E,
                  smem + (size_t)cur * BUF_E + 3 * TILE_E,
                  lane, wm, wn, acc);
    }
#else
    // fallback: synchronous cooperative copy, single buffered
    for (int kk = 0; kk < nk; ++kk) {
        const int k0 = kk * BK;
        __syncthreads();
        for (int e = tid * 4; e < BM * BK; e += 256 * 4) {
            int row = e >> 5, col = e & 31;
            *(us4*)(smem + row * LDT + col)               = *(const us4*)(Xhi + (size_t)(m0 + row) * D_DIM + k0 + col);
            *(us4*)(smem + TILE_E + row * LDT + col)      = *(const us4*)(Xlo + (size_t)(m0 + row) * D_DIM + k0 + col);
            *(us4*)(smem + 2 * TILE_E + row * LDT + col)  = *(const us4*)(Whi + (size_t)(f0 + row) * D_DIM + k0 + col);
            *(us4*)(smem + 3 * TILE_E + row * LDT + col)  = *(const us4*)(Wlo + (size_t)(f0 + row) * D_DIM + k0 + col);
        }
        __syncthreads();
        wmma_step(smem, smem + TILE_E, smem + 2 * TILE_E, smem + 3 * TILE_E, lane, wm, wn, acc);
    }
#endif

    // epilogue: store project tile + per-block sum/sumsq partials
    float s = 0.f, s2 = 0.f;
    #pragma unroll
    for (int mi = 0; mi < 2; ++mi) {
        #pragma unroll
        for (int ni = 0; ni < 4; ++ni) {
            int gm = m0 + wm + mi * 16 + ((lane < 16) ? 0 : 8);   // C: vgpr r -> M=r or 8+r
            int gn = f0 + wn + ni * 16 + (lane & 15);
            #pragma unroll
            for (int r = 0; r < 8; ++r) {
                float v = acc[mi][ni][r];
                project[(size_t)(gm + r) * F_DIM + gn] = v;
                s += v; s2 += v * v;
            }
        }
    }
    __syncthreads();
    float* red = (float*)smem;   // reuse LDS (512 floats)
    red[tid] = s; red[256 + tid] = s2;
    __syncthreads();
    for (int ofs = 128; ofs > 0; ofs >>= 1) {
        if (tid < ofs) { red[tid] += red[tid + ofs]; red[256 + tid] += red[256 + tid + ofs]; }
        __syncthreads();
    }
    if (tid == 0) {
        int bid = blockIdx.y * gridDim.x + blockIdx.x;
        blocksums[2 * bid]     = (double)red[0];
        blocksums[2 * bid + 1] = (double)red[256];
    }
}

// =====================================================================
// Kernel 2: fixed-order reduction of block partials -> fuzz scale
// =====================================================================
__global__ __launch_bounds__(256)
void finalize_std_kernel(const double* __restrict__ blocksums, int nblocks,
                         float* __restrict__ fuzz_scale)
{
    __shared__ double sr[256][2];
    int tid = threadIdx.x;
    double s = 0.0, s2 = 0.0;
    for (int i = tid; i < nblocks; i += 256) { s += blocksums[2 * i]; s2 += blocksums[2 * i + 1]; }
    sr[tid][0] = s; sr[tid][1] = s2;
    __syncthreads();
    for (int ofs = 128; ofs > 0; ofs >>= 1) {
        if (tid < ofs) { sr[tid][0] += sr[tid + ofs][0]; sr[tid][1] += sr[tid + ofs][1]; }
        __syncthreads();
    }
    if (tid == 0) {
        double N = (double)B_DIM * (double)F_DIM;
        double mean = sr[0][0] / N;
        double var  = (sr[0][1] - sr[0][0] * mean) / (N - 1.0);   // ddof=1
        *fuzz_scale = FUZZ_FACTOR * (float)sqrt(var);
    }
}

// =====================================================================
// Kernel 3: per-row top-k (threshold bisection) + both decodes
// =====================================================================
__device__ __forceinline__ int block_count_gt(const float* __restrict__ buf, float t,
                                              int tid, int* s_redi)
{
    int c = 0;
    for (int i = tid; i < F_DIM; i += 256) c += (buf[i] > t) ? 1 : 0;
    for (int d = 16; d > 0; d >>= 1) c += __shfl_down(c, d);
    if ((tid & 31) == 0) s_redi[tid >> 5] = c;
    __syncthreads();
    int tot = 0;
    #pragma unroll
    for (int j = 0; j < 8; ++j) tot += s_redi[j];
    __syncthreads();
    return tot;
}

__device__ __forceinline__ float2 threshold_search(const float* __restrict__ buf, int K,
                                                   int tid, int* s_redi, float* s_redf)
{
    float mx = -3.0e38f, mn = 3.0e38f;
    for (int i = tid; i < F_DIM; i += 256) {
        float v = buf[i];
        if (v > -1.0e37f) { mx = fmaxf(mx, v); mn = fminf(mn, v); }
    }
    for (int d = 16; d > 0; d >>= 1) { mx = fmaxf(mx, __shfl_down(mx, d)); mn = fminf(mn, __shfl_down(mn, d)); }
    if ((tid & 31) == 0) { s_redf[tid >> 5] = mx; s_redf[8 + (tid >> 5)] = mn; }
    __syncthreads();
    #pragma unroll
    for (int j = 0; j < 8; ++j) { mx = fmaxf(mx, s_redf[j]); mn = fminf(mn, s_redf[8 + j]); }
    __syncthreads();

    float lo = mn - (fabsf(mn) * 1e-3f + 1e-3f);
    float hi = mx;
    for (int it = 0; it < 44; ++it) {
        float t = 0.5f * (lo + hi);
        int c = block_count_gt(buf, t, tid, s_redi);
        if (c >= K) lo = t; else hi = t;
    }
    return make_float2(lo, hi);
}

__device__ __forceinline__ int collect_gt(const float* __restrict__ buf, float t, int cap,
                                          int tid, int* s_scan, int* out_f, float* out_w)
{
    int c = 0;
    for (int i = tid; i < F_DIM; i += 256) c += (buf[i] > t) ? 1 : 0;
    s_scan[tid] = c;
    __syncthreads();
    for (int ofs = 1; ofs < 256; ofs <<= 1) {
        int v = s_scan[tid];
        int add = (tid >= ofs) ? s_scan[tid - ofs] : 0;
        __syncthreads();
        s_scan[tid] = v + add;
        __syncthreads();
    }
    int pos = s_scan[tid] - c;
    int total = s_scan[255];
    __syncthreads();
    for (int i = tid; i < F_DIM; i += 256) {
        float v = buf[i];
        if (v > t) {
            if (pos < cap) { out_f[pos] = i; out_w[pos] = v; }
            ++pos;
        }
    }
    __syncthreads();
    return total;
}

__device__ __forceinline__ void fill_boundary(const float* __restrict__ buf, float lo, float hi,
                                              int K, int n1, int tid, int* s_scan,
                                              int* cand_i, float* cand_v,
                                              int* out_f, float* out_w)
{
    int c = 0;
    for (int i = tid; i < F_DIM; i += 256) { float v = buf[i]; c += (v > lo && v <= hi) ? 1 : 0; }
    s_scan[tid] = c;
    __syncthreads();
    for (int ofs = 1; ofs < 256; ofs <<= 1) {
        int v = s_scan[tid];
        int add = (tid >= ofs) ? s_scan[tid - ofs] : 0;
        __syncthreads();
        s_scan[tid] = v + add;
        __syncthreads();
    }
    int pos = s_scan[tid] - c;
    int total = s_scan[255];
    __syncthreads();
    for (int i = tid; i < F_DIM; i += 256) {
        float v = buf[i];
        if (v > lo && v <= hi) {
            if (pos < 64) { cand_i[pos] = i; cand_v[pos] = v; }
            ++pos;
        }
    }
    __syncthreads();
    if (tid == 0) {
        int need = K - n1;
        int nc = (total < 64) ? total : 64;
        for (int k = 0; k < need && k < nc; ++k) {
            int bj = 0, bidx = 0x7FFFFFFF;
            for (int j = 0; j < nc; ++j)
                if (cand_i[j] < bidx) { bidx = cand_i[j]; bj = j; }
            out_f[n1 + k] = cand_i[bj];
            out_w[n1 + k] = cand_v[bj];
            cand_i[bj] = 0x7FFFFFFF;
        }
    }
    __syncthreads();
}

__global__ __launch_bounds__(256)
void topk_decode_kernel(const float* __restrict__ project,
                        const float* __restrict__ enc_bias,
                        const float* __restrict__ lookup,
                        const float* __restrict__ noise,
                        const int*   __restrict__ last_usage,
                        const float* __restrict__ fuzz_scale,
                        float* __restrict__ out)   // [2*B*D] : recon | undead
{
    extern __shared__ float buf[];                // F_DIM floats (96 KB)
    __shared__ float s_w[DEAD_TOPK];
    __shared__ int   s_f[DEAD_TOPK];
    __shared__ int   s_scan[256];
    __shared__ int   s_redi[8];
    __shared__ float s_redf[16];
    __shared__ int   s_cand_i[64];
    __shared__ float s_cand_v[64];

    const int b   = blockIdx.x;
    const int tid = threadIdx.x;
    const size_t bF = (size_t)b * F_DIM;

    for (int i = tid; i < F_DIM; i += 256) buf[i] = project[bF + i];
    __syncthreads();

    // ---- top-32 ----
    float2 th = threshold_search(buf, TOPK, tid, s_redi, s_redf);
    int n1 = collect_gt(buf, th.y, TOPK, tid, s_scan, s_f, s_w);
    fill_boundary(buf, th.x, th.y, TOPK, n1, tid, s_scan, s_cand_i, s_cand_v, s_f, s_w);

    // ---- decode: embed_recon = sum_k w_k * lookup[f_k] + enc_bias ----
    {
        int col = tid * 4;
        float4 a = *(const float4*)(enc_bias + col);
        #pragma unroll 4
        for (int k = 0; k < TOPK; ++k) {
            float w = s_w[k];
            const float4 lv = *(const float4*)(lookup + (size_t)s_f[k] * D_DIM + col);
            a.x += w * lv.x; a.y += w * lv.y; a.z += w * lv.z; a.w += w * lv.w;
        }
        *(float4*)(out + (size_t)b * D_DIM + col) = a;
    }
    __syncthreads();

    // ---- dead-feature fuzz transform (in place) ----
    float fs = *fuzz_scale;
    for (int i = tid; i < F_DIM; i += 256) {
        float v = buf[i];
        bool dead = last_usage[i] > DEAD_CUTOFF;
        buf[i] = dead ? (v + fs * noise[bF + i]) : -__builtin_huge_valf();
    }
    __syncthreads();

    // ---- top-256 over dead_project ----
    th = threshold_search(buf, DEAD_TOPK, tid, s_redi, s_redf);
    n1 = collect_gt(buf, th.y, DEAD_TOPK, tid, s_scan, s_f, s_w);
    fill_boundary(buf, th.x, th.y, DEAD_TOPK, n1, tid, s_scan, s_cand_i, s_cand_v, s_f, s_w);

    if (tid < DEAD_TOPK) s_w[tid] = project[bF + s_f[tid]];  // weights = ORIGINAL project
    __syncthreads();

    // ---- decode: undead_recon (no bias) ----
    {
        int col = tid * 4;
        float4 a = { 0.f, 0.f, 0.f, 0.f };
        #pragma unroll 4
        for (int k = 0; k < DEAD_TOPK; ++k) {
            float w = s_w[k];
            const float4 lv = *(const float4*)(lookup + (size_t)s_f[k] * D_DIM + col);
            a.x += w * lv.x; a.y += w * lv.y; a.z += w * lv.z; a.w += w * lv.w;
        }
        *(float4*)(out + (size_t)(B_DIM + b) * D_DIM + col) = a;
    }
}

// =====================================================================
extern "C" void kernel_launch(void* const* d_in, const int* in_sizes, int n_in,
                              void* d_out, int out_size, void* d_ws, size_t ws_size,
                              hipStream_t stream)
{
    const float* embed      = (const float*)d_in[0];
    const float* enc_bias   = (const float*)d_in[1];
    const float* enc_W      = (const float*)d_in[2];
    const float* lookup     = (const float*)d_in[3];
    const float* noise      = (const float*)d_in[4];
    const int*   last_usage = (const int*)d_in[5];
    float* out = (float*)d_out;

    // workspace layout (~520 MB): project | blocksums | fuzz | Xhi | Xlo | Whi | Wlo
    char* ws = (char*)d_ws;
    size_t off = 0;
    float* project = (float*)(ws + off);          off += (size_t)B_DIM * F_DIM * sizeof(float);
    const int nblocks = (F_DIM / BN) * (B_DIM / BM);
    double* blocksums = (double*)(ws + off);      off += (size_t)nblocks * 2 * sizeof(double);
    float* fuzz_scale = (float*)(ws + off);       off += 16;
    unsigned short* Xhi = (unsigned short*)(ws + off); off += (size_t)B_DIM * D_DIM * 2;
    unsigned short* Xlo = (unsigned short*)(ws + off); off += (size_t)B_DIM * D_DIM * 2;
    unsigned short* Whi = (unsigned short*)(ws + off); off += (size_t)F_DIM * D_DIM * 2;
    unsigned short* Wlo = (unsigned short*)(ws + off); off += (size_t)F_DIM * D_DIM * 2;

    split_kernel<<<2048, 256, 0, stream>>>(embed, enc_bias, Xhi, Xlo, B_DIM);
    split_kernel<<<4096, 256, 0, stream>>>(enc_W, nullptr, Whi, Wlo, F_DIM);

    dim3 g1(F_DIM / BN, B_DIM / BM);          // 192 x 32
    gemm_project_kernel<<<g1, 256, GEMM_SMEM_BYTES, stream>>>(Xhi, Xlo, Whi, Wlo, project, blocksums);
    finalize_std_kernel<<<1, 256, 0, stream>>>(blocksums, nblocks, fuzz_scale);
    topk_decode_kernel<<<B_DIM, 256, F_DIM * sizeof(float), stream>>>(
        project, enc_bias, lookup, noise, last_usage, fuzz_scale, out);
}
